// GFSA_Attn_Timm_41704132444732
// MI455X (gfx1250) — compile-verified, matrix operand order checked by reading, not run on hardware
//
#include <hip/hip_runtime.h>

// ---------------------------------------------------------------------------
// GFSA attention (timm-style) for MI455X / gfx1250.
// GEMMs via v_wmma_f32_16x16x32_bf16; the dominant attn@attn kernel stages
// operand panels into LDS with the Tensor Data Mover (tensor_load_to_lds,
// TENSORcnt double-buffered) so WMMA feeds from ds_load_b128.
// ---------------------------------------------------------------------------

typedef __bf16 bf16;
typedef __attribute__((ext_vector_type(8)))  __bf16 v8bf;
typedef __attribute__((ext_vector_type(16))) __bf16 v16bf;
typedef __attribute__((ext_vector_type(8)))  float  v8f;
typedef __attribute__((ext_vector_type(4)))  unsigned uint32x4;
typedef __attribute__((ext_vector_type(8)))  int      int32x8;
typedef __attribute__((ext_vector_type(4)))  int      int32x4;

#define BATCH   4
#define SEQ     1024
#define DIM     768
#define NHEADS  12
#define HDIM    64
#define BH      (BATCH * NHEADS)          // 48
#define MROWS   (BATCH * SEQ)             // 4096
#define QKVDIM  (3 * DIM)                 // 2304
#define NCB     16                        // column 64-blocks of the NxN matrix

__device__ __forceinline__ v8f wmma_bf16(v16bf a, v16bf b, v8f c) {
  return __builtin_amdgcn_wmma_f32_16x16x32_bf16(
      /*neg_a=*/false, a, /*neg_b=*/false, b,
      /*c_mod=*/(short)0, c, /*reuse_a=*/false, /*reuse_b=*/false);
}

// Load one 16x32 bf16 operand tile in the ISA per-lane layout (16-bit A
// matrix 16x32): lane<16 holds K {0..7,16..23}, lane>=16 holds K {8..15,
// 24..31} of row M=lane&15.  Works for global or LDS pointers.
__device__ __forceinline__ v16bf load_tile_kmajor(const bf16* __restrict__ base,
                                                  int ld, int rc, int kb, int lane) {
  int m  = lane & 15;
  int ko = (lane & 16) ? 8 : 0;
  const bf16* p = base + (size_t)(rc + m) * ld + kb + ko;
  v8bf lo = *(const v8bf*)(p);
  v8bf hi = *(const v8bf*)(p + 16);
  return __builtin_shufflevector(lo, hi, 0,1,2,3,4,5,6,7,8,9,10,11,12,13,14,15);
}

__device__ __forceinline__ v8f vzero8() {
  v8f z = {0.f,0.f,0.f,0.f,0.f,0.f,0.f,0.f};
  return z;
}

// ---- Tensor Data Mover: 2D tile load (64 rows x 32 bf16, row stride SEQ) ---
// D# packing per CDNA5 ISA §8.3/8.4: group0 {count=1 | lds_addr | global_addr
// | type=2}; group1 {data_size=2B, tensor 1024x1024, tile 32(K)x64(rows),
// dim0_stride=1024}; groups 2/3 zero (2D tile).
__device__ __forceinline__ void tdm_load_64x32(unsigned lds_off, const bf16* gsrc) {
  unsigned long long ga = (unsigned long long)(size_t)gsrc;
  uint32x4 g0 = { 1u,                                   // count=1, user mode
                  lds_off,                              // LDS byte address
                  (unsigned)ga,                         // global_addr[31:0]
                  (unsigned)(ga >> 32) | (2u << 30) };  // addr[56:32] | type=2
  int32x8 g1 = { (int)(1u << 16),                       // data_size=1 (2 bytes)
                 (int)(((unsigned)SEQ & 0xFFFFu) << 16),        // dim0[15:0]
                 (int)((((unsigned)SEQ >> 16) & 0xFFFFu)
                       | (((unsigned)SEQ & 0xFFFFu) << 16)),    // dim0 hi|dim1 lo
                 (int)((((unsigned)SEQ >> 16) & 0xFFFFu)
                       | (32u << 16)),                          // dim1 hi|tile_dim0=32
                 (int)64u,                              // tile_dim1=64, tile_dim2=0
                 (int)SEQ,                              // tensor_dim0_stride lo32
                 0, 0 };                                // stride hi / dim1_stride
  int32x4 gz = { 0, 0, 0, 0 };
#if defined(__clang_major__) && __clang_major__ >= 23
  int32x8 z8 = { 0, 0, 0, 0, 0, 0, 0, 0 };
  __builtin_amdgcn_tensor_load_to_lds(g0, g1, gz, gz, z8, 0);
#else
  __builtin_amdgcn_tensor_load_to_lds(g0, g1, gz, gz, 0);
#endif
}

__device__ __forceinline__ unsigned lds_offset_of(const void* p) {
  // generic pointers into the LDS aperture carry the LDS byte offset in the
  // low 32 bits
  return (unsigned)(size_t)p;
}

// --------------------------- stage 0: converts ------------------------------
__global__ void k_f32_to_bf16(const float* __restrict__ src, bf16* __restrict__ dst, int n) {
  int i = blockIdx.x * 256 + threadIdx.x;
  if (i < n) dst[i] = (bf16)src[i];
}

// --------------------------- stage 1: QKV GEMM ------------------------------
__global__ __launch_bounds__(128) void k_gemm_qkv(const bf16* __restrict__ xb,
                                                  const bf16* __restrict__ wb,
                                                  const float* __restrict__ bias,
                                                  bf16* __restrict__ qd,
                                                  bf16* __restrict__ kd,
                                                  bf16* __restrict__ vtd) {
  int lane = threadIdx.x & 31, wv = threadIdx.x >> 5;
  int tm  = blockIdx.x * 64 + wv * 16;
  int tn0 = blockIdx.y * 64;
  v8f acc[4];
  #pragma unroll
  for (int t = 0; t < 4; ++t) acc[t] = vzero8();

  for (int kb = 0; kb < DIM; kb += 32) {
    v16bf a = load_tile_kmajor(xb, DIM, tm, kb, lane);
    #pragma unroll
    for (int t = 0; t < 4; ++t) {
      v16bf b = load_tile_kmajor(wb, DIM, tn0 + t * 16, kb, lane);
      acc[t] = wmma_bf16(a, b, acc[t]);
    }
  }

  #pragma unroll
  for (int t = 0; t < 4; ++t) {
    int cn = tn0 + t * 16 + (lane & 15);
    float bv = bias[cn];
    int which = cn / DIM;
    int rem   = cn % DIM;
    int h = rem >> 6, d = rem & 63;
    #pragma unroll
    for (int r = 0; r < 8; ++r) {
      int rg = tm + r + ((lane >> 4) << 3);
      int bi = rg >> 10, n = rg & 1023;
      float v = acc[t][r] + bv;
      size_t hoff = (((size_t)bi * NHEADS + h) * SEQ + n) * HDIM + d;
      if (which == 0)      qd[hoff] = (bf16)v;
      else if (which == 1) kd[hoff] = (bf16)v;
      else vtd[(((size_t)bi * NHEADS + h) * HDIM + d) * SEQ + n] = (bf16)v;
    }
  }
}

// ----------------- stage 2: scores (WMMA) + softmax (fp32 LDS) --------------
__global__ __launch_bounds__(128) void k_scores_softmax(const bf16* __restrict__ q,
                                                        const bf16* __restrict__ k,
                                                        bf16* __restrict__ attn,
                                                        bf16* __restrict__ attn_t) {
  extern __shared__ float S[];           // 16*1024 f32 panel
  float* red   = S + 16 * 1024;          // [8 segs][16 rows]
  float* rstat = red + 128;              // [16 rows]
  int lane = threadIdx.x & 31, wv = threadIdx.x >> 5;
  int tm = blockIdx.x * 16;
  int bh = blockIdx.y;
  const bf16* qb = q + (size_t)bh * SEQ * HDIM;
  const bf16* kp = k + (size_t)bh * SEQ * HDIM;
  const float scale = 0.125f;            // HDIM^-0.5

  for (int t = 0; t < 16; ++t) {
    int ct = wv * 16 + t;
    v8f acc = vzero8();
    #pragma unroll
    for (int kk = 0; kk < HDIM; kk += 32) {
      v16bf a = load_tile_kmajor(qb, HDIM, tm, kk, lane);
      v16bf b = load_tile_kmajor(kp, HDIM, ct * 16, kk, lane);
      acc = wmma_bf16(a, b, acc);
    }
    int col = ct * 16 + (lane & 15);
    #pragma unroll
    for (int r = 0; r < 8; ++r) {
      int row = r + ((lane >> 4) << 3);
      S[row * 1024 + col] = acc[r] * scale;
    }
  }
  __syncthreads();

  int row = threadIdx.x & 15;
  int seg = threadIdx.x >> 4;
  int c0  = seg * 128;

  float m = -3.4e38f;
  for (int c = 0; c < 128; ++c) m = fmaxf(m, S[row * 1024 + c0 + c]);
  red[seg * 16 + row] = m;
  __syncthreads();
  if (threadIdx.x < 16) {
    float mm = red[row];
    for (int s = 1; s < 8; ++s) mm = fmaxf(mm, red[s * 16 + row]);
    rstat[row] = mm;
  }
  __syncthreads();
  float rm = rstat[row];
  float ls = 0.f;
  for (int c = 0; c < 128; ++c) {
    float e = __expf(S[row * 1024 + c0 + c] - rm);
    S[row * 1024 + c0 + c] = e;
    ls += e;
  }
  red[seg * 16 + row] = ls;
  __syncthreads();
  if (threadIdx.x < 16) {
    float ss = 0.f;
    for (int s = 0; s < 8; ++s) ss += red[s * 16 + row];
    rstat[row] = 1.0f / ss;
  }
  __syncthreads();
  float inv = rstat[row];
  size_t aoff = (size_t)bh << 20;
  int rg = tm + row;
  for (int c = 0; c < 128; ++c) {
    int cg = c0 + c;
    bf16 pb = (bf16)(S[row * 1024 + cg] * inv);
    attn  [aoff + (size_t)rg * 1024 + cg] = pb;   // row-major (A operand)
    attn_t[aoff + (size_t)cg * 1024 + rg] = pb;   // transposed (B operand)
  }
}

// -------- stage 3: attn@attn via TDM-staged LDS panels + GFSA mix -----------
// Block = 64 rows x 64 cols; wave0 DMAs the 64x32 A and B K-slices into LDS
// (double buffered, TENSORcnt), all 4 waves consume them via ds_load_b128.
// final = delta*I + (gamma-6*lamb)*attn + 7*lamb*(attn@attn), ReLU, partial
// row sums written to unique slots (deterministic renorm).
__global__ __launch_bounds__(128) void k_square_mix(const bf16* __restrict__ attn,
                                                    const bf16* __restrict__ attn_t,
                                                    bf16* __restrict__ attnF,
                                                    float* __restrict__ rsum_part,
                                                    const float* __restrict__ lamb,
                                                    const float* __restrict__ gamma,
                                                    const float* __restrict__ delta) {
  __shared__ bf16 sA[2][64 * 32];        // 4 KB per buffer
  __shared__ bf16 sB[2][64 * 32];
  int lane = threadIdx.x & 31, wv = threadIdx.x >> 5;
  int bh = blockIdx.z, h = bh % NHEADS;
  int tmB = blockIdx.y * 64;             // block row base
  int tm  = tmB + wv * 16;               // this wave's rows
  int cb  = blockIdx.x;
  int tn0 = cb * 64;
  size_t aoff = (size_t)bh << 20;
  const bf16* A  = attn   + aoff;
  const bf16* Bt = attn_t + aoff;

  v8f acc[4];
  #pragma unroll
  for (int t = 0; t < 4; ++t) acc[t] = vzero8();

  if (wv == 0) {                         // prologue: stage K-slice 0
    tdm_load_64x32(lds_offset_of(&sA[0][0]), A  + (size_t)tmB * SEQ);
    tdm_load_64x32(lds_offset_of(&sB[0][0]), Bt + (size_t)tn0 * SEQ);
  }

  for (int it = 0; it < SEQ / 32; ++it) {
    int cur = it & 1, nxt = cur ^ 1;
    if (wv == 0) {
      if (it + 1 < SEQ / 32) {
        int kb2 = (it + 1) * 32;
        tdm_load_64x32(lds_offset_of(&sA[nxt][0]), A  + (size_t)tmB * SEQ + kb2);
        tdm_load_64x32(lds_offset_of(&sB[nxt][0]), Bt + (size_t)tn0 * SEQ + kb2);
        __builtin_amdgcn_s_wait_tensorcnt(2);  // in-order: current pair done
      } else {
        __builtin_amdgcn_s_wait_tensorcnt(0);
      }
    }
    __syncthreads();                     // staged slice visible to all waves
    v16bf a = load_tile_kmajor(&sA[cur][0], 32, wv * 16, 0, lane);
    #pragma unroll
    for (int t = 0; t < 4; ++t) {
      v16bf b = load_tile_kmajor(&sB[cur][0], 32, t * 16, 0, lane);
      acc[t] = wmma_bf16(a, b, acc[t]);
    }
    __syncthreads();                     // done reading before buffer reuse
  }

  float lam = lamb[h];
  float c1 = gamma[h] - 6.0f * lam;
  float c2 = 7.0f * lam;
  float dl = delta[h];

  float rs[8];
  #pragma unroll
  for (int r = 0; r < 8; ++r) rs[r] = 0.f;

  #pragma unroll
  for (int t = 0; t < 4; ++t) {
    int col = tn0 + t * 16 + (lane & 15);
    #pragma unroll
    for (int r = 0; r < 8; ++r) {
      int rowg = tm + r + ((lane >> 4) << 3);
      float av = (float)A[(size_t)rowg * 1024 + col];
      float f  = c2 * acc[t][r] + c1 * av;
      if (rowg == col) f += dl;
      f = fmaxf(f, 0.0f);
      attnF[aoff + (size_t)rowg * 1024 + col] = (bf16)f;
      rs[r] += f;
    }
  }
  #pragma unroll
  for (int r = 0; r < 8; ++r) {
    float v = rs[r];
    v += __shfl_xor(v, 1);
    v += __shfl_xor(v, 2);
    v += __shfl_xor(v, 4);
    v += __shfl_xor(v, 8);
    if ((lane & 15) == 0) {
      int rowg = tm + r + ((lane >> 4) << 3);
      rsum_part[(size_t)cb * (BH * SEQ) + bh * SEQ + rowg] = v;
    }
  }
}

// ------------------- stage 4: (attnF/rowsum) @ v  (WMMA) --------------------
__global__ __launch_bounds__(128) void k_av(const bf16* __restrict__ attnF,
                                            const bf16* __restrict__ vt,
                                            const float* __restrict__ rsum_part,
                                            bf16* __restrict__ outb) {
  int lane = threadIdx.x & 31, wv = threadIdx.x >> 5;
  int bh = blockIdx.y, bi = bh / NHEADS, h = bh % NHEADS;
  int tm = blockIdx.x * 64 + wv * 16;
  size_t aoff = (size_t)bh << 20;
  const bf16* A  = attnF + aoff;
  const bf16* Vt = vt + (size_t)bh * HDIM * SEQ;

  float dinv[8];
  #pragma unroll
  for (int r = 0; r < 8; ++r) {
    int n = tm + r + ((lane >> 4) << 3);
    float s = 1e-9f;
    for (int cb = 0; cb < NCB; ++cb)
      s += rsum_part[(size_t)cb * (BH * SEQ) + bh * SEQ + n];
    dinv[r] = 1.0f / s;
  }

  v8f acc[4];
  #pragma unroll
  for (int t = 0; t < 4; ++t) acc[t] = vzero8();

  for (int kb = 0; kb < SEQ; kb += 32) {
    v16bf a = load_tile_kmajor(A, SEQ, tm, kb, lane);
    __builtin_prefetch(A + (size_t)(tm + (lane & 15)) * SEQ + kb + 32, 0, 1);
    #pragma unroll
    for (int t = 0; t < 4; ++t) {
      v16bf b = load_tile_kmajor(Vt, SEQ, t * 16, kb, lane);
      acc[t] = wmma_bf16(a, b, acc[t]);
    }
  }

  #pragma unroll
  for (int t = 0; t < 4; ++t) {
    int d = t * 16 + (lane & 15);
    #pragma unroll
    for (int r = 0; r < 8; ++r) {
      int n = tm + r + ((lane >> 4) << 3);
      float o = acc[t][r] * dinv[r];
      outb[((size_t)bi * SEQ + n) * DIM + h * HDIM + d] = (bf16)o;
    }
  }
}

// ----------------------- stage 5: projection GEMM ---------------------------
__global__ __launch_bounds__(128) void k_proj(const bf16* __restrict__ outb,
                                              const bf16* __restrict__ pw,
                                              const float* __restrict__ pb,
                                              float* __restrict__ y) {
  int lane = threadIdx.x & 31, wv = threadIdx.x >> 5;
  int tm  = blockIdx.x * 64 + wv * 16;
  int tn0 = blockIdx.y * 64;
  v8f acc[4];
  #pragma unroll
  for (int t = 0; t < 4; ++t) acc[t] = vzero8();

  for (int kb = 0; kb < DIM; kb += 32) {
    v16bf a = load_tile_kmajor(outb, DIM, tm, kb, lane);
    #pragma unroll
    for (int t = 0; t < 4; ++t) {
      v16bf b = load_tile_kmajor(pw, DIM, tn0 + t * 16, kb, lane);
      acc[t] = wmma_bf16(a, b, acc[t]);
    }
  }

  #pragma unroll
  for (int t = 0; t < 4; ++t) {
    int cn = tn0 + t * 16 + (lane & 15);
    float bv = pb[cn];
    #pragma unroll
    for (int r = 0; r < 8; ++r) {
      int rg = tm + r + ((lane >> 4) << 3);
      y[(size_t)rg * DIM + cn] = acc[t][r] + bv;
    }
  }
}

// ---------------------------------------------------------------------------
extern "C" void kernel_launch(void* const* d_in, const int* in_sizes, int n_in,
                              void* d_out, int out_size, void* d_ws, size_t ws_size,
                              hipStream_t stream) {
  const float* x      = (const float*)d_in[0];
  const float* qkv_w  = (const float*)d_in[1];
  const float* qkv_b  = (const float*)d_in[2];
  const float* proj_w = (const float*)d_in[3];
  const float* proj_b = (const float*)d_in[4];
  const float* lamb   = (const float*)d_in[5];
  const float* gamma  = (const float*)d_in[6];
  const float* delta  = (const float*)d_in[7];
  float* y = (float*)d_out;

  char* w = (char*)d_ws;
  size_t o = 0;
  auto alloc = [&](size_t bytes) -> char* {
    char* p = w + o;
    o += (bytes + 255) & ~(size_t)255;
    return p;
  };
  bf16*  x_bf    = (bf16*) alloc((size_t)MROWS * DIM * 2);
  bf16*  qkvw_bf = (bf16*) alloc((size_t)QKVDIM * DIM * 2);
  bf16*  projw_bf= (bf16*) alloc((size_t)DIM * DIM * 2);
  bf16*  q_bf    = (bf16*) alloc((size_t)BH * SEQ * HDIM * 2);
  bf16*  k_bf    = (bf16*) alloc((size_t)BH * SEQ * HDIM * 2);
  bf16*  v_t     = (bf16*) alloc((size_t)BH * HDIM * SEQ * 2);
  bf16*  attn    = (bf16*) alloc((size_t)BH * SEQ * SEQ * 2);
  bf16*  attn_t  = (bf16*) alloc((size_t)BH * SEQ * SEQ * 2);
  bf16*  attnF   = (bf16*) alloc((size_t)BH * SEQ * SEQ * 2);
  float* rs_part = (float*)alloc((size_t)NCB * BH * SEQ * 4);
  bf16*  out_bf  = (bf16*) alloc((size_t)MROWS * DIM * 2);
  (void)ws_size; (void)in_sizes; (void)n_in; (void)out_size;

  {
    int n0 = MROWS * DIM;
    k_f32_to_bf16<<<(n0 + 255) / 256, 256, 0, stream>>>(x, x_bf, n0);
    int n1 = QKVDIM * DIM;
    k_f32_to_bf16<<<(n1 + 255) / 256, 256, 0, stream>>>(qkv_w, qkvw_bf, n1);
    int n2 = DIM * DIM;
    k_f32_to_bf16<<<(n2 + 255) / 256, 256, 0, stream>>>(proj_w, projw_bf, n2);
  }
  k_gemm_qkv<<<dim3(MROWS / 64, QKVDIM / 64), 128, 0, stream>>>(
      x_bf, qkvw_bf, qkv_b, q_bf, k_bf, v_t);
  size_t shmem = (16 * 1024 + 128 + 16) * sizeof(float);
  k_scores_softmax<<<dim3(SEQ / 16, BH), 128, shmem, stream>>>(
      q_bf, k_bf, attn, attn_t);
  k_square_mix<<<dim3(NCB, SEQ / 64, BH), 128, 0, stream>>>(
      attn, attn_t, attnF, rs_part, lamb, gamma, delta);
  k_av<<<dim3(SEQ / 64, BH), 128, 0, stream>>>(attnF, v_t, rs_part, out_bf);
  k_proj<<<dim3(MROWS / 64, DIM / 64), 128, 0, stream>>>(
      out_bf, projw_bf, proj_b, y);
}
